// MistralMoLoraLayer_71081708748822
// MI455X (gfx1250) — matrix operand order repaired and checked by reading, not compiled, not run on hardware
//
#include <hip/hip_runtime.h>
#include <math.h>

typedef __attribute__((ext_vector_type(2)))  float  v2f;
typedef __attribute__((ext_vector_type(4)))  float  v4f;
typedef __attribute__((ext_vector_type(8)))  float  v8f;
typedef __attribute__((ext_vector_type(8)))  __bf16 v8bf;
typedef __attribute__((ext_vector_type(16))) __bf16 v16bf;
typedef __attribute__((ext_vector_type(4)))  unsigned short v4us;

#define DD 2048      // hidden dim == in dim
#define NE 8         // experts
#define NR 16        // lora rank
#define NK 2         // top-k
#define LALPHA 2.0f

// ---------------------------------------------------------------- bf16 helpers
__device__ __forceinline__ __bf16 f2bf(float f) {
    unsigned u = __builtin_bit_cast(unsigned, f);
    unsigned short s = (unsigned short)((u + 0x7FFFu + ((u >> 16) & 1u)) >> 16); // RNE
    return __builtin_bit_cast(__bf16, s);
}
__device__ __forceinline__ float bf2f(__bf16 b) {
    unsigned u = ((unsigned)__builtin_bit_cast(unsigned short, b)) << 16;
    return __builtin_bit_cast(float, u);
}

// ---------------------------------------------------------------- WMMA helper
// D = A(16x32,bf16) * B(32x16,bf16) + C(16x16,f32), wave32.
// A frag per lane (m = lane%16, h = lane/16): elems j=0..15 = A[m][k0+16h+j]
// B frag per lane (n = lane%16, h = lane/16): elems j=0..15 = B[k0+16h+j][n]
// C per lane: c[i] -> row m = i + 8*(lane/16), col n = lane%16
__device__ __forceinline__ v8f wmma_bf16(v16bf a, v16bf b, v8f c) {
    return __builtin_amdgcn_wmma_f32_16x16x32_bf16(false, a, false, b,
                                                   (short)0, c, false, false);
}
__device__ __forceinline__ v16bf load_a_frag(const __bf16* p) {
    v8bf c0 = *(const v8bf*)(p);
    v8bf c1 = *(const v8bf*)(p + 8);
    return __builtin_shufflevector(c0, c1, 0,1,2,3,4,5,6,7,8,9,10,11,12,13,14,15);
}

// ---------------------------------------------------------------- fp32 -> hi/lo bf16
__global__ void __launch_bounds__(256)
split_kernel(const float* __restrict__ src, __bf16* __restrict__ hi,
             __bf16* __restrict__ lo, int n)
{
    int i = (blockIdx.x * blockDim.x + threadIdx.x) * 4;
    if (i >= n) return;
    v4f v = *(const v4f*)(src + i);
    v4us h, l;
#pragma unroll
    for (int j = 0; j < 4; ++j) {
        __bf16 hb = f2bf(v[j]);
        __bf16 lb = f2bf(v[j] - bf2f(hb));
        h[j] = __builtin_bit_cast(unsigned short, hb);
        l[j] = __builtin_bit_cast(unsigned short, lb);
    }
    *(v4us*)((unsigned short*)hi + i) = h;
    *(v4us*)((unsigned short*)lo + i) = l;
}

// ---------------------------------------------------------------- router
__global__ void __launch_bounds__(256)
router_kernel(const float* __restrict__ x, const float* __restrict__ Wg,
              int* __restrict__ sel, float* __restrict__ wgt, int T)
{
    int wave = (blockIdx.x * blockDim.x + threadIdx.x) >> 5;
    int lane = threadIdx.x & 31;
    if (wave >= T) return;
    const float* xr = x + (size_t)wave * DD;
    float acc[NE];
#pragma unroll
    for (int e = 0; e < NE; ++e) acc[e] = 0.f;
    for (int d = lane; d < DD; d += 32) {
        float xv = xr[d];
#pragma unroll
        for (int e = 0; e < NE; ++e)
            acc[e] = fmaf(xv, Wg[e * DD + d], acc[e]);
    }
#pragma unroll
    for (int e = 0; e < NE; ++e)
        for (int off = 16; off > 0; off >>= 1)
            acc[e] += __shfl_xor(acc[e], off, 32);
    if (lane == 0) {
        int i0 = 0; float v0 = acc[0];
        for (int e = 1; e < NE; ++e) if (acc[e] > v0) { v0 = acc[e]; i0 = e; }
        int i1 = (i0 == 0) ? 1 : 0; float v1 = acc[i1];
        for (int e = 0; e < NE; ++e)
            if (e != i0 && acc[e] > v1) { v1 = acc[e]; i1 = e; }
        float r = __expf(v1 - v0);          // softmax over the 2 selected logits
        float w0 = 1.0f / (1.0f + r);
        sel[wave * 2 + 0] = i0; sel[wave * 2 + 1] = i1;
        wgt[wave * 2 + 0] = w0; wgt[wave * 2 + 1] = 1.0f - w0;
    }
}

// ---------------------------------------------------------------- rank-16 input proj
// p1[t,k,j] = dot(x[t,:], B1[sel[t,k]][j,:]);  p3 likewise from B3.
__global__ void __launch_bounds__(64)
lora_in_kernel(const float* __restrict__ x,
               const float* __restrict__ B1, const float* __restrict__ B3,
               const int* __restrict__ sel,
               float* __restrict__ p1, float* __restrict__ p3)
{
    int t   = blockIdx.x;
    int tid = threadIdx.x;
    int k   = tid >> 5;
    int mat = (tid >> 4) & 1;
    int j   = tid & 15;
    int e   = sel[t * NK + k];
    const float* B  = (mat ? B3 : B1) + ((size_t)e * NR + j) * DD;
    const float* xr = x + (size_t)t * DD;
    float acc = 0.f;
    for (int d = 0; d < DD; d += 4) {
        v4f xv = *(const v4f*)(xr + d);
        v4f bv = *(const v4f*)(B + d);
        acc = fmaf(xv.x, bv.x, fmaf(xv.y, bv.y, fmaf(xv.z, bv.z, fmaf(xv.w, bv.w, acc))));
    }
    (mat ? p3 : p1)[(t * NK + k) * NR + j] = acc;
}

// ---------------------------------------------------------------- split-bf16 WMMA GEMM
// C[M,2048] = (Ahi+Alo)[M,2048] @ (Whi+Wlo).T  via  AhiBhi + AloBhi + AhiBlo
// 256 threads = 8 waves arranged 2M x 4N; wave tile 32x64; block tile 64 x 256.
// Per K=32 step per wave: 24 b128 loads for 24 WMMAs (B frags reused over 2 M-tiles).
__global__ void __launch_bounds__(256)
gemm_split_kernel(const __bf16* __restrict__ Ahi, const __bf16* __restrict__ Alo,
                  const __bf16* __restrict__ Whi, const __bf16* __restrict__ Wlo,
                  float* __restrict__ C)
{
    int lane = threadIdx.x & 31;
    int w    = threadIdx.x >> 5;
    int wm   = w >> 2;                 // 0..1
    int wn   = w & 3;                  // 0..3
    int m0   = blockIdx.y * 64 + wm * 32;
    int n0   = blockIdx.x * 256 + wn * 64;
    int l16  = lane & 15;
    int half = lane >> 4;

    const __bf16* ah0 = Ahi + (size_t)(m0 + l16) * DD + 16 * half;
    const __bf16* al0 = Alo + (size_t)(m0 + l16) * DD + 16 * half;
    const __bf16* ah1 = ah0 + (size_t)16 * DD;
    const __bf16* al1 = al0 + (size_t)16 * DD;
    const __bf16* wh  = Whi + (size_t)(n0 + l16) * DD + 16 * half;
    const __bf16* wl  = Wlo + (size_t)(n0 + l16) * DD + 16 * half;

    v8f acc[2][4];
#pragma unroll
    for (int mt = 0; mt < 2; ++mt)
#pragma unroll
        for (int j = 0; j < 4; ++j)
            acc[mt][j] = (v8f){0.f,0.f,0.f,0.f,0.f,0.f,0.f,0.f};

#pragma unroll 1
    for (int k0 = 0; k0 < DD; k0 += 32) {
        v16bf ahf0 = load_a_frag(ah0 + k0);
        v16bf alf0 = load_a_frag(al0 + k0);
        v16bf ahf1 = load_a_frag(ah1 + k0);
        v16bf alf1 = load_a_frag(al1 + k0);
#pragma unroll
        for (int j = 0; j < 4; ++j) {
            size_t noff = (size_t)(16 * j) * DD + k0;
            v16bf bhf = *(const v16bf*)(wh + noff);
            v16bf blf = *(const v16bf*)(wl + noff);
            acc[0][j] = wmma_bf16(ahf0, bhf, acc[0][j]);
            acc[1][j] = wmma_bf16(ahf1, bhf, acc[1][j]);
            acc[0][j] = wmma_bf16(alf0, bhf, acc[0][j]);
            acc[1][j] = wmma_bf16(alf1, bhf, acc[1][j]);
            acc[0][j] = wmma_bf16(ahf0, blf, acc[0][j]);
            acc[1][j] = wmma_bf16(ahf1, blf, acc[1][j]);
        }
    }

#pragma unroll
    for (int mt = 0; mt < 2; ++mt) {
        int rbase = m0 + mt * 16 + half * 8;
#pragma unroll
        for (int i = 0; i < 8; ++i) {
            size_t row = (size_t)(rbase + i) * DD;
            C[row + n0 +  0 + l16] = acc[mt][0][i];
            C[row + n0 + 16 + l16] = acc[mt][1][i];
            C[row + n0 + 32 + l16] = acc[mt][2][i];
            C[row + n0 + 48 + l16] = acc[mt][3][i];
        }
    }
}

// ---------------------------------------------------------------- h = silu(h1)*h3 (writes hi/lo bf16)
__global__ void __launch_bounds__(256)
h_kernel(const float* __restrict__ U, const float* __restrict__ G,
         const float* __restrict__ A1, const float* __restrict__ A3,
         const float* __restrict__ p1, const float* __restrict__ p3,
         const int* __restrict__ sel,
         __bf16* __restrict__ Hhi, __bf16* __restrict__ Hlo)
{
    int r = blockIdx.y;                         // token-expert slot: r = t*2 + k
    int d = blockIdx.x * blockDim.x + threadIdx.x;
    int t = r >> 1;
    int e = sel[r];
    const float* p1r = p1 + r * NR;
    const float* p3r = p3 + r * NR;
    const float* a1  = A1 + ((size_t)e * DD + d) * NR;
    const float* a3  = A3 + ((size_t)e * DD + d) * NR;
    float s1 = 0.f, s3 = 0.f;
#pragma unroll
    for (int j = 0; j < NR; ++j) {
        s1 = fmaf(p1r[j], a1[j], s1);
        s3 = fmaf(p3r[j], a3[j], s3);
    }
    float h1 = U[(size_t)t * DD + d] + LALPHA * s1;
    float h3 = G[(size_t)t * DD + d] + LALPHA * s3;
    float sig = 1.0f / (1.0f + __expf(-h1));
    float hv  = (h1 * sig) * h3;
    __bf16 hb = f2bf(hv);
    Hhi[(size_t)r * DD + d] = hb;
    Hlo[(size_t)r * DD + d] = f2bf(hv - bf2f(hb));
}

// ---------------------------------------------------------------- rank-16 down proj
// p2[r,j] = dot(H[r,:], B2[sel[r]][j,:]) with H reconstructed from hi+lo
__global__ void __launch_bounds__(256)
lora2_kernel(const __bf16* __restrict__ Hhi, const __bf16* __restrict__ Hlo,
             const float* __restrict__ B2, const int* __restrict__ sel,
             float* __restrict__ p2)
{
    int idx = blockIdx.x * blockDim.x + threadIdx.x;
    int r = idx >> 4;
    int j = idx & 15;
    int e = sel[r];
    const float* B = B2 + ((size_t)e * NR + j) * DD;
    const __bf16* hh = Hhi + (size_t)r * DD;
    const __bf16* hl = Hlo + (size_t)r * DD;
    float acc = 0.f;
    for (int d = 0; d < DD; d += 8) {
        v8bf vh = *(const v8bf*)(hh + d);
        v8bf vl = *(const v8bf*)(hl + d);
#pragma unroll
        for (int q = 0; q < 8; ++q)
            acc = fmaf(bf2f(vh[q]) + bf2f(vl[q]), B[d + q], acc);
    }
    p2[r * NR + j] = acc;
}

// ---------------------------------------------------------------- combine epilogue
// out[t,n] = sum_k wgt[r] * (O[r,n] + alpha * dot16(p2[r,:], A2[e][n,:])), r = t*2+k
__global__ void __launch_bounds__(256)
finalize_kernel(const float* __restrict__ O, const float* __restrict__ A2,
                const float* __restrict__ p2, const int* __restrict__ sel,
                const float* __restrict__ wgt, float* __restrict__ out)
{
    int t = blockIdx.y;
    int n = blockIdx.x * blockDim.x + threadIdx.x;
    float res = 0.f;
#pragma unroll
    for (int k = 0; k < NK; ++k) {
        int r = t * NK + k;
        int e = sel[r];
        const float* a2 = A2 + ((size_t)e * DD + n) * NR;
        const float* pr = p2 + r * NR;
        float s = 0.f;
#pragma unroll
        for (int j = 0; j < NR; ++j) s = fmaf(pr[j], a2[j], s);
        res += wgt[r] * (O[(size_t)r * DD + n] + LALPHA * s);
    }
    out[(size_t)t * DD + n] = res;
}

// ---------------------------------------------------------------- launch
extern "C" void kernel_launch(void* const* d_in, const int* in_sizes, int n_in,
                              void* d_out, int out_size, void* d_ws, size_t ws_size,
                              hipStream_t stream)
{
    const float* x     = (const float*)d_in[0];
    const float* Wg    = (const float*)d_in[1];
    const float* Wup   = (const float*)d_in[2];
    const float* Wgate = (const float*)d_in[3];
    const float* Wdown = (const float*)d_in[4];
    const float* A1    = (const float*)d_in[5];
    const float* B1    = (const float*)d_in[6];
    const float* A2    = (const float*)d_in[7];
    const float* B2    = (const float*)d_in[8];
    const float* A3    = (const float*)d_in[9];
    const float* B3    = (const float*)d_in[10];
    float* out = (float*)d_out;
    const int T = in_sizes[0] / DD;     // 8192

    // workspace carve-out (256B aligned)
    char* ws = (char*)d_ws;
    size_t off = 0;
    auto carve = [&](size_t bytes) -> char* {
        char* p = ws + off;
        off += (bytes + 255) & ~(size_t)255;
        return p;
    };
    int*    sel  = (int*)   carve(sizeof(int)    * (size_t)T * NK);
    float*  wgt  = (float*) carve(sizeof(float)  * (size_t)T * NK);
    float*  p1   = (float*) carve(sizeof(float)  * (size_t)T * NK * NR);
    float*  p3   = (float*) carve(sizeof(float)  * (size_t)T * NK * NR);
    float*  p2   = (float*) carve(sizeof(float)  * (size_t)T * NK * NR);
    __bf16* xhi  = (__bf16*)carve(sizeof(__bf16) * (size_t)T * DD);
    __bf16* xlo  = (__bf16*)carve(sizeof(__bf16) * (size_t)T * DD);
    __bf16* Wuh  = (__bf16*)carve(sizeof(__bf16) * (size_t)DD * DD);
    __bf16* Wul  = (__bf16*)carve(sizeof(__bf16) * (size_t)DD * DD);
    __bf16* Wgh  = (__bf16*)carve(sizeof(__bf16) * (size_t)DD * DD);
    __bf16* Wgl  = (__bf16*)carve(sizeof(__bf16) * (size_t)DD * DD);
    __bf16* Wdh  = (__bf16*)carve(sizeof(__bf16) * (size_t)DD * DD);
    __bf16* Wdl  = (__bf16*)carve(sizeof(__bf16) * (size_t)DD * DD);
    float*  U    = (float*) carve(sizeof(float)  * (size_t)T * DD);
    float*  G    = (float*) carve(sizeof(float)  * (size_t)T * DD);
    __bf16* Hlo  = (__bf16*)carve(sizeof(__bf16) * (size_t)T * NK * DD);
    // aliases over dead regions:
    __bf16* Hhi  = xhi;      // x splits (2*T*DD bf16) dead after U/G GEMMs
    float*  O    = U;        // U+G (2*T*DD f32) dead after h_kernel

    // ---- precision splits
    split_kernel<<<(T * DD / 4) / 256, 256, 0, stream>>>(x, xhi, xlo, T * DD);
    split_kernel<<<(DD * DD / 4) / 256, 256, 0, stream>>>(Wup,   Wuh, Wul, DD * DD);
    split_kernel<<<(DD * DD / 4) / 256, 256, 0, stream>>>(Wgate, Wgh, Wgl, DD * DD);
    split_kernel<<<(DD * DD / 4) / 256, 256, 0, stream>>>(Wdown, Wdh, Wdl, DD * DD);

    // ---- router + input LoRA
    router_kernel<<<T / 8, 256, 0, stream>>>(x, Wg, sel, wgt, T);
    lora_in_kernel<<<T, 64, 0, stream>>>(x, B1, B3, sel, p1, p3);

    // ---- shared up/gate GEMMs (block tile 64 x 256)
    dim3 gs(DD / 256, T / 64);
    gemm_split_kernel<<<gs, 256, 0, stream>>>(xhi, xlo, Wuh, Wul, U);
    gemm_split_kernel<<<gs, 256, 0, stream>>>(xhi, xlo, Wgh, Wgl, G);

    // ---- per-(token,expert) activation (consumes x splits' region as Hhi)
    dim3 gh(DD / 256, T * NK);
    h_kernel<<<gh, 256, 0, stream>>>(U, G, A1, A3, p1, p3, sel, Hhi, Hlo);

    // ---- down LoRA + down GEMM (O overwrites dead U/G)
    lora2_kernel<<<(T * NK * NR) / 256, 256, 0, stream>>>(Hhi, Hlo, B2, sel, p2);
    dim3 gd(DD / 256, (T * NK) / 64);
    gemm_split_kernel<<<gd, 256, 0, stream>>>(Hhi, Hlo, Wdh, Wdl, O);

    // ---- combine
    dim3 gf(DD / 256, T);
    finalize_kernel<<<gf, 256, 0, stream>>>(O, A2, p2, sel, wgt, out);
}
